// SelfAttention_20538533610167
// MI455X (gfx1250) — compile-verified, matrix operand order checked by reading, not run on hardware
//
#include <hip/hip_runtime.h>
#include <stdint.h>

typedef __attribute__((ext_vector_type(16))) _Float16 v16h;
typedef __attribute__((ext_vector_type(8)))  _Float16 v8h;
typedef __attribute__((ext_vector_type(8)))  float    v8f;

#define BATCH 4
#define SEQ   2048
#define EMB   1024
#define HEADS 16
#define HDIM  64
#define NEGVAL (-1.0e7f)

// ---------------------------------------------------------------------------
// Fragment helpers for v_wmma_f32_16x16x32_f16 (wave32, gfx1250 layouts).
// A-frag (16x32 MxK) / B-frag loaded from an N-major [rows x ld] f16 matrix:
//   lane 0-15:  row = base+lane,     K = {0..7, 16..23}
//   lane 16-31: row = base+lane-16,  K = {8..15, 24..31}
// Each lane: two contiguous 16-byte chunks -> two b128 loads.
// ---------------------------------------------------------------------------
__device__ __forceinline__ v16h load_frag_rowmajor(const _Float16* __restrict__ src,
                                                   int rowBase, int k0, int ld) {
    const int lane = threadIdx.x & 31;
    const int lrow = lane & 15;
    const int kba  = (lane >> 4) << 3;           // 0 or 8
    const _Float16* p = src + (size_t)(rowBase + lrow) * (size_t)ld + k0 + kba;
    v8h c0 = *(const v8h*)(p);                   // K = kba .. kba+7
    v8h c1 = *(const v8h*)(p + 16);              // K = kba+16 .. kba+23
    return __builtin_shufflevector(c0, c1, 0,1,2,3,4,5,6,7,8,9,10,11,12,13,14,15);
}

// Same fragment pattern but from an LDS tile (compiles to ds_load_b128).
__device__ __forceinline__ v16h load_frag_lds(const _Float16* base,
                                              int rowBase, int k0, int ld) {
    const int lane = threadIdx.x & 31;
    const int lrow = lane & 15;
    const int kba  = (lane >> 4) << 3;
    const _Float16* p = base + (rowBase + lrow) * ld + k0 + kba;
    v8h c0 = *(const v8h*)(p);
    v8h c1 = *(const v8h*)(p + 16);
    return __builtin_shufflevector(c0, c1, 0,1,2,3,4,5,6,7,8,9,10,11,12,13,14,15);
}

__device__ __forceinline__ v8f wmma_f16(v16h a, v16h b, v8f c) {
    // (neg_a, A, neg_b, B, c_mod, C, reuse_a, reuse_b)
    return __builtin_amdgcn_wmma_f32_16x16x32_f16(false, a, false, b, (short)0, c,
                                                  false, false);
}

// ---------------------------------------------------------------------------
// CDNA5 async memory->LDS copy (16 bytes per lane), tracked by ASYNCcnt.
// Low 32 bits of a generic pointer to __shared__ are the LDS byte offset.
// ---------------------------------------------------------------------------
__device__ __forceinline__ void async_copy_b128(void* lds_ptr, const void* gptr) {
    unsigned lds_addr = (unsigned)(uintptr_t)lds_ptr;
    unsigned long long ga = (unsigned long long)(uintptr_t)gptr;
    asm volatile("global_load_async_to_lds_b128 %0, %1, off"
                 :: "v"(lds_addr), "v"(ga)
                 : "memory");
}

__device__ __forceinline__ void wait_async0() {
#if __has_builtin(__builtin_amdgcn_s_wait_asynccnt)
    __builtin_amdgcn_s_wait_asynccnt(0);
#else
    asm volatile("s_wait_asynccnt 0x0" ::: "memory");
#endif
}

// ---------------------------------------------------------------------------
// fp32 -> fp16 conversion
// ---------------------------------------------------------------------------
__global__ void cvt_f32_to_f16_kernel(const float* __restrict__ src,
                                      _Float16* __restrict__ dst, int n) {
    int i = blockIdx.x * blockDim.x + threadIdx.x;
    if (i < n) dst[i] = (_Float16)src[i];
}

// ---------------------------------------------------------------------------
// QKV projection: C[m][n] = sum_k X[m][k] * W[n][k]   (x @ W.T)
// M = b*t = 8192, N = e = 1024, K = e = 1024.
// Block: 256 threads = 8 waves, block tile 128x128, wave tile 64Mx32N.
// g = blockIdx.z selects Q (scaled by e^-1/2), K, or V (stored transposed).
// ---------------------------------------------------------------------------
__global__ __launch_bounds__(256) void qkv_proj_kernel(
    const _Float16* __restrict__ xh,
    const _Float16* __restrict__ Wqh, const _Float16* __restrict__ Wkh,
    const _Float16* __restrict__ Wvh,
    _Float16* __restrict__ Qh, _Float16* __restrict__ Kh,
    _Float16* __restrict__ Vth)
{
    const int g = blockIdx.z;
    const _Float16* Wh = (g == 0) ? Wqh : (g == 1) ? Wkh : Wvh;
    const int w    = threadIdx.x >> 5;
    const int wm   = w & 1, wn = w >> 1;
    const int m0   = blockIdx.x * 128 + wm * 64;
    const int n0   = blockIdx.y * 128 + wn * 32;
    const int lane = threadIdx.x & 31;
    const int lrow = lane & 15, lhi = lane >> 4;

    v8f acc[4][2];
    #pragma unroll
    for (int i = 0; i < 4; i++)
        #pragma unroll
        for (int j = 0; j < 2; j++)
            acc[i][j] = (v8f){0.f,0.f,0.f,0.f,0.f,0.f,0.f,0.f};

    for (int k0 = 0; k0 < EMB; k0 += 32) {
        v16h a[4], bf[2];
        #pragma unroll
        for (int i = 0; i < 4; i++) a[i]  = load_frag_rowmajor(xh, m0 + 16*i, k0, EMB);
        #pragma unroll
        for (int j = 0; j < 2; j++) bf[j] = load_frag_rowmajor(Wh, n0 + 16*j, k0, EMB);
        #pragma unroll
        for (int i = 0; i < 4; i++)
            #pragma unroll
            for (int j = 0; j < 2; j++)
                acc[i][j] = wmma_f16(a[i], bf[j], acc[i][j]);
    }

    const float qscale = 0.03125f;               // EMB^(-1/2): net q*k scaling
    #pragma unroll
    for (int i = 0; i < 4; i++)
        #pragma unroll
        for (int j = 0; j < 2; j++)
            #pragma unroll
            for (int r = 0; r < 8; r++) {
                int row  = m0 + 16*i + 8*lhi + r;     // global token row [0,8192)
                int col  = n0 + 16*j + lrow;          // feature [0,1024)
                int bb   = row >> 11, tok = row & (SEQ - 1);
                int head = col >> 6,  dim = col & (HDIM - 1);
                float v = acc[i][j][r];
                if (g == 0) {
                    Qh[((size_t)(bb*HEADS + head)*SEQ + tok)*HDIM + dim] =
                        (_Float16)(v * qscale);
                } else if (g == 1) {
                    Kh[((size_t)(bb*HEADS + head)*SEQ + tok)*HDIM + dim] = (_Float16)v;
                } else {
                    Vth[((size_t)(bb*HEADS + head)*HDIM + dim)*SEQ + tok] = (_Float16)v;
                }
            }
}

// ---------------------------------------------------------------------------
// Flash attention: grid (t/64, heads, batch), 128 threads = 4 waves,
// each wave owns 16 query rows. Online softmax over key tiles of 32.
// K/V tiles are staged ONCE per block into LDS via async-to-LDS copies
// (double buffered), shared by all 4 waves.
// ---------------------------------------------------------------------------
__global__ __launch_bounds__(128) void flash_attn_kernel(
    const _Float16* __restrict__ Qh, const _Float16* __restrict__ Kh,
    const _Float16* __restrict__ Vth, const uint8_t* __restrict__ mask,
    _Float16* __restrict__ Oh)
{
    __shared__ __align__(16) _Float16 ldsP[4][16][32];   // per-wave P tile, 4KB
    __shared__ __align__(16) _Float16 ldsK[2][32][HDIM]; // 2 x 4KB  [key][dim]
    __shared__ __align__(16) _Float16 ldsV[2][HDIM][32]; // 2 x 4KB  [dim][key]

    const int tid  = threadIdx.x;
    const int w    = tid >> 5;
    const int lane = tid & 31;
    const int lrow = lane & 15, lhi = lane >> 4;
    const int kba  = lhi << 3;
    const int b    = blockIdx.z, head = blockIdx.y;
    const int q0   = blockIdx.x * 64 + w * 16;

    const _Float16* Qp = Qh  + (size_t)(b*HEADS + head) * SEQ * HDIM;
    const _Float16* Kp = Kh  + (size_t)(b*HEADS + head) * SEQ * HDIM;
    const _Float16* Vp = Vth + (size_t)(b*HEADS + head) * HDIM * SEQ; // [dim][tok]
    const uint8_t*  mr = mask + (size_t)b * SEQ;

    // Stage one 32-key K tile (4KB, contiguous) + V tile (64 rows x 64B) into
    // LDS buffer `buf`. 128 threads x 2 chunks x 16B each per tile.
    auto stage_tile = [&](int buf, int kt) {
        #pragma unroll
        for (int i = 0; i < 2; i++) {
            int c = tid + 128 * i;                         // chunk 0..255
            async_copy_b128(((_Float16*)&ldsK[buf][0][0]) + c * 8,
                            Kp + (size_t)kt * HDIM + c * 8);
        }
        #pragma unroll
        for (int i = 0; i < 2; i++) {
            int c   = tid + 128 * i;                       // chunk 0..255
            int row = c >> 2, col = c & 3;                 // row=dim, 4x16B/row
            async_copy_b128(&ldsV[buf][row][col * 8],
                            Vp + (size_t)row * SEQ + kt + col * 8);
        }
    };

    v16h aq[2];
    aq[0] = load_frag_rowmajor(Qp, q0, 0,  HDIM);
    aq[1] = load_frag_rowmajor(Qp, q0, 32, HDIM);

    v8f o[4];
    #pragma unroll
    for (int j = 0; j < 4; j++) o[j] = (v8f){0.f,0.f,0.f,0.f,0.f,0.f,0.f,0.f};
    float mx[8], ls[8];
    #pragma unroll
    for (int r = 0; r < 8; r++) { mx[r] = -3.0e30f; ls[r] = 0.0f; }

    int buf = 0;
    stage_tile(0, 0);
    wait_async0();
    __syncthreads();

    for (int kt = 0; kt < SEQ; kt += 32) {
        if (kt + 32 < SEQ) stage_tile(buf ^ 1, kt + 32);   // overlap next tile

        // ---- S = Q . K^T for 32 keys: two 16x16 score fragments (from LDS K)
        v8f s0 = (v8f){0.f,0.f,0.f,0.f,0.f,0.f,0.f,0.f};
        v8f s1 = (v8f){0.f,0.f,0.f,0.f,0.f,0.f,0.f,0.f};
        #pragma unroll
        for (int kk = 0; kk < 2; kk++) {
            v16h bk0 = load_frag_lds(&ldsK[buf][0][0], 0,  kk*32, HDIM);
            v16h bk1 = load_frag_lds(&ldsK[buf][0][0], 16, kk*32, HDIM);
            s0 = wmma_f16(aq[kk], bk0, s0);
            s1 = wmma_f16(aq[kk], bk1, s1);
        }
        const bool ok0 = mr[kt + lrow]      != 0;
        const bool ok1 = mr[kt + 16 + lrow] != 0;

        // ---- online softmax (rows live across the 16-lane half-wave)
        float c[8];
        #pragma unroll
        for (int r = 0; r < 8; r++) {
            float a0 = ok0 ? (float)s0[r] : NEGVAL;
            float a1 = ok1 ? (float)s1[r] : NEGVAL;
            float rm = fmaxf(a0, a1);
            rm = fmaxf(rm, __shfl_xor(rm, 1, 32));
            rm = fmaxf(rm, __shfl_xor(rm, 2, 32));
            rm = fmaxf(rm, __shfl_xor(rm, 4, 32));
            rm = fmaxf(rm, __shfl_xor(rm, 8, 32));
            float mnew = fmaxf(mx[r], rm);
            float cr = __expf(mx[r] - mnew);
            float p0 = __expf(a0 - mnew);
            float p1 = __expf(a1 - mnew);
            mx[r] = mnew;
            ls[r] = ls[r] * cr + p0 + p1;        // lane-local partial row sum
            c[r]  = cr;
            ldsP[w][8*lhi + r][lrow]      = (_Float16)p0;
            ldsP[w][8*lhi + r][lrow + 16] = (_Float16)p1;
        }
        #pragma unroll
        for (int j = 0; j < 4; j++)
            #pragma unroll
            for (int r = 0; r < 8; r++) o[j][r] *= c[r];

        __syncthreads();
        // ---- re-layout P (C-layout -> A-fragment) via LDS
        v8h pc0 = *(const v8h*)&ldsP[w][lrow][kba];
        v8h pc1 = *(const v8h*)&ldsP[w][lrow][kba + 16];
        v16h pa = __builtin_shufflevector(pc0, pc1,
                     0,1,2,3,4,5,6,7,8,9,10,11,12,13,14,15);

        // ---- O += P . V  (B-frags from LDS V, [dim][key])
        #pragma unroll
        for (int j = 0; j < 4; j++) {
            v16h bv = load_frag_lds(&ldsV[buf][0][0], 16*j, 0, 32);
            o[j] = wmma_f16(pa, bv, o[j]);
        }

        wait_async0();       // next tile landed in LDS (this wave's copies)
        __syncthreads();     // all waves done with buf + P, next buffer ready
        buf ^= 1;
    }

    // ---- epilogue: normalize and store f16 O in [b, t, e] layout
    #pragma unroll
    for (int r = 0; r < 8; r++) {
        float l = ls[r];
        l += __shfl_xor(l, 1, 32);
        l += __shfl_xor(l, 2, 32);
        l += __shfl_xor(l, 4, 32);
        l += __shfl_xor(l, 8, 32);
        float inv = 1.0f / l;
        int q = q0 + 8*lhi + r;
        #pragma unroll
        for (int j = 0; j < 4; j++) {
            int dim = 16*j + lrow;
            Oh[(size_t)(b*SEQ + q)*EMB + head*HDIM + dim] =
                (_Float16)((float)o[j][r] * inv);
        }
    }
}

// ---------------------------------------------------------------------------
// Output projection: out = O @ Wu.T + bu, fp32 store.
// ---------------------------------------------------------------------------
__global__ __launch_bounds__(256) void out_proj_kernel(
    const _Float16* __restrict__ Oh, const _Float16* __restrict__ Wuh,
    const float* __restrict__ bu, float* __restrict__ out)
{
    const int w    = threadIdx.x >> 5;
    const int wm   = w & 1, wn = w >> 1;
    const int m0   = blockIdx.x * 128 + wm * 64;
    const int n0   = blockIdx.y * 128 + wn * 32;
    const int lane = threadIdx.x & 31;
    const int lrow = lane & 15, lhi = lane >> 4;

    v8f acc[4][2];
    #pragma unroll
    for (int i = 0; i < 4; i++)
        #pragma unroll
        for (int j = 0; j < 2; j++)
            acc[i][j] = (v8f){0.f,0.f,0.f,0.f,0.f,0.f,0.f,0.f};

    for (int k0 = 0; k0 < EMB; k0 += 32) {
        v16h a[4], bf[2];
        #pragma unroll
        for (int i = 0; i < 4; i++) a[i]  = load_frag_rowmajor(Oh,  m0 + 16*i, k0, EMB);
        #pragma unroll
        for (int j = 0; j < 2; j++) bf[j] = load_frag_rowmajor(Wuh, n0 + 16*j, k0, EMB);
        #pragma unroll
        for (int i = 0; i < 4; i++)
            #pragma unroll
            for (int j = 0; j < 2; j++)
                acc[i][j] = wmma_f16(a[i], bf[j], acc[i][j]);
    }

    #pragma unroll
    for (int i = 0; i < 4; i++)
        #pragma unroll
        for (int j = 0; j < 2; j++)
            #pragma unroll
            for (int r = 0; r < 8; r++) {
                int row = m0 + 16*i + 8*lhi + r;
                int col = n0 + 16*j + lrow;
                out[(size_t)row * EMB + col] = acc[i][j][r] + bu[col];
            }
}

// ---------------------------------------------------------------------------
// Host launcher
// ---------------------------------------------------------------------------
extern "C" void kernel_launch(void* const* d_in, const int* in_sizes, int n_in,
                              void* d_out, int out_size, void* d_ws, size_t ws_size,
                              hipStream_t stream) {
    const float*   x    = (const float*)d_in[0];
    const uint8_t* mask = (const uint8_t*)d_in[1];
    const float*   Wq   = (const float*)d_in[2];
    const float*   Wk   = (const float*)d_in[3];
    const float*   Wv   = (const float*)d_in[4];
    const float*   Wu   = (const float*)d_in[5];
    const float*   bu   = (const float*)d_in[6];
    float*         out  = (float*)d_out;

    const size_t NX = (size_t)BATCH * SEQ * EMB;   // 8,388,608
    const size_t NW = (size_t)EMB * EMB;           // 1,048,576

    char* ws = (char*)d_ws;
    size_t off = 0;
    auto carve = [&](size_t bytes) {
        char* p = ws + off;
        off += (bytes + 255) & ~(size_t)255;
        return p;
    };
    _Float16* xh  = (_Float16*)carve(NX * 2);
    _Float16* Wqh = (_Float16*)carve(NW * 2);
    _Float16* Wkh = (_Float16*)carve(NW * 2);
    _Float16* Wvh = (_Float16*)carve(NW * 2);
    _Float16* Wuh = (_Float16*)carve(NW * 2);
    _Float16* Qh  = (_Float16*)carve(NX * 2);
    _Float16* Kh  = (_Float16*)carve(NX * 2);
    _Float16* Vth = (_Float16*)carve(NX * 2);
    _Float16* Ohp = (_Float16*)carve(NX * 2);
    (void)ws_size; (void)in_sizes; (void)n_in; (void)out_size;

    const int cb = 256;
    cvt_f32_to_f16_kernel<<<(int)((NX + cb - 1) / cb), cb, 0, stream>>>(x,  xh,  (int)NX);
    cvt_f32_to_f16_kernel<<<(int)((NW + cb - 1) / cb), cb, 0, stream>>>(Wq, Wqh, (int)NW);
    cvt_f32_to_f16_kernel<<<(int)((NW + cb - 1) / cb), cb, 0, stream>>>(Wk, Wkh, (int)NW);
    cvt_f32_to_f16_kernel<<<(int)((NW + cb - 1) / cb), cb, 0, stream>>>(Wv, Wvh, (int)NW);
    cvt_f32_to_f16_kernel<<<(int)((NW + cb - 1) / cb), cb, 0, stream>>>(Wu, Wuh, (int)NW);

    qkv_proj_kernel<<<dim3(64, 8, 3), 256, 0, stream>>>(xh, Wqh, Wkh, Wvh, Qh, Kh, Vth);

    flash_attn_kernel<<<dim3(SEQ / 64, HEADS, BATCH), 128, 0, stream>>>(
        Qh, Kh, Vth, mask, Ohp);

    out_proj_kernel<<<dim3(64, 8), 256, 0, stream>>>(Ohp, Wuh, bu, out);
}
